// GNNForceField_19739669692447
// MI455X (gfx1250) — matrix operand order repaired and checked by reading, not compiled, not run on hardware
//
#include <hip/hip_runtime.h>

// ---------------------------------------------------------------------------
// GNN force field for MI455X (gfx1250, wave32, WMMA).
// Edge MLP dominates (114 GFLOP, ~4.5 GB HBM traffic -> memory bound).
// This revision:
//  * edge-MLP weights staged once per block into LDS (kills ~8 GB of repeated
//    L2 B-fragment reads; hot loop now ds_load_b128 + v_wmma)
//  * double-buffered per-wave A tiles filled with GLOBAL_LOAD_ASYNC_TO_LDS_B128
//    (ASYNCcnt) so random h16 gathers overlap WMMA compute
//  * float4 scatter, global_prefetch for streamed ea
// ---------------------------------------------------------------------------

#define NN   25000
#define NE   800000
#define IND  16
#define HID  128
#define EDIM 50
#define EP   64      // EDIM padded to 64
#define NL   4
#define KE   320     // 2*HID + EP (edge-MLP K, padded from 306)
#define TPW  5       // edge tiles per wave (NE = 2500 blocks * 4 waves * 5 * 16)

typedef _Float16 half_t;
typedef __attribute__((ext_vector_type(16))) _Float16 v16h;
typedef __attribute__((ext_vector_type(8)))  _Float16 v8h;
typedef __attribute__((ext_vector_type(8)))  float    v8f;

__device__ __forceinline__ float silu_f(float x) {
    return x * (1.0f / (1.0f + __expf(-x)));
}

__device__ __forceinline__ v8f wmma_f16(v16h a, v16h b, v8f c) {
    return __builtin_amdgcn_wmma_f32_16x16x32_f16(false, a, false, b,
                                                  (short)0, c, false, false);
}

// A fragment (16x32 f16, row-major tile in LDS, stride in halves).
// Lane m<16 holds K{0..7,16..23}; lane m+16 holds K{8..15,24..31}.
__device__ __forceinline__ v16h a_frag_lds(const half_t* Abase, int stride,
                                           int kt, int lane) {
    int m  = lane & 15;
    int k0 = kt * 32 + ((lane >> 4) << 3);
    const half_t* p = Abase + m * stride + k0;
    v8h lo = *(const v8h*)(p);
    v8h hi = *(const v8h*)(p + 16);
    return __builtin_shufflevector(lo, hi, 0, 1, 2, 3, 4, 5, 6, 7,
                                   8, 9, 10, 11, 12, 13, 14, 15);
}

// B fragment: pre-packed so each lane reads 16 contiguous halves (32B).
__device__ __forceinline__ v16h b_frag(const half_t* __restrict__ W,
                                       int ktiles, int nt, int kt, int lane) {
    return *(const v16h*)(W + ((size_t)((nt * ktiles + kt) * 32 + lane)) * 16);
}

// --- gfx1250 async copy: per-lane global -> per-lane LDS, tracked by ASYNCcnt.
__device__ __forceinline__ void async_b128(unsigned lds_off, const half_t* g) {
    asm volatile("global_load_async_to_lds_b128 %0, %1, off"
                 :: "v"(lds_off), "v"(g) : "memory");
}
__device__ __forceinline__ void wait_async_le16() {
    asm volatile("s_wait_asynccnt 0x10" ::: "memory");
}
__device__ __forceinline__ void wait_async_0() {
    asm volatile("s_wait_asynccnt 0x0" ::: "memory");
}

// --- Weight packing: f32 row-major [Ksrc][Nsrc] -> f16 B-fragment order,
//     zero padded to [Kpad][Npad].
__global__ void pack_b_kernel(const float* __restrict__ src, half_t* __restrict__ dst,
                              int Ksrc, int Nsrc, int Kpad, int Npad) {
    int ktiles = Kpad / 32;
    long total = (long)Kpad * Npad;
    long i = (long)blockIdx.x * 256 + threadIdx.x;
    if (i >= total) return;
    int  j    = (int)(i & 15);
    long t    = i >> 4;
    int  lane = (int)(t & 31);  t >>= 5;
    int  kt   = (int)(t % ktiles);
    int  nt   = (int)(t / ktiles);
    int  n = (nt << 4) + (lane & 15);
    int  k = kt * 32 + ((lane >> 4) << 3) + (j & 7) + ((j >> 3) << 4);
    float v = (k < Ksrc && n < Nsrc) ? src[(size_t)k * Nsrc + n] : 0.0f;
    dst[i] = (half_t)v;
}

// --- Node embedding: h = x @ emb_w + emb_b  (and f16 shadow)
__global__ void embed_kernel(const float* __restrict__ x, const float* __restrict__ w,
                             const float* __restrict__ b, float* __restrict__ h,
                             half_t* __restrict__ h16) {
    long i = (long)blockIdx.x * 256 + threadIdx.x;
    if (i >= (long)NN * HID) return;
    int n = (int)(i / HID), d = (int)(i % HID);
    float acc = b[d];
#pragma unroll
    for (int k = 0; k < IND; k++) acc += x[(size_t)n * IND + k] * w[(size_t)k * HID + d];
    h[i]   = acc;
    h16[i] = (half_t)acc;
}

// --- Gaussian smearing into padded [E][EP] f32 (pad cols stay 0)
__global__ void smear_kernel(const float* __restrict__ eattr, float* __restrict__ ea) {
    long i = (long)blockIdx.x * 256 + threadIdx.x;
    if (i >= (long)NE * EP) return;
    int  d = (int)(i & (EP - 1));
    long e = i >> 6;
    float v = 0.0f;
    if (d < EDIM) {
        const float step  = 5.0f / (EDIM - 1);
        const float gamma = -0.5f / (step * step);
        float diff = eattr[e] - step * (float)d;
        v = __expf(gamma * diff * diff);
    }
    ea[i] = v;
}

__global__ void zero_kernel(float* __restrict__ p, long n) {
    long i = (long)blockIdx.x * 256 + threadIdx.x;
    if (i < n) p[i] = 0.0f;
}

// --- agg[col] += h[row] : float4 read + 4 f32 atomics per thread
__global__ void scatter_kernel(const float* __restrict__ h, const int* __restrict__ row,
                               const int* __restrict__ col, float* __restrict__ agg) {
    long i = (long)blockIdx.x * 256 + threadIdx.x;     // NE * 32 threads
    if (i >= (long)NE * 32) return;
    int e = (int)(i >> 5);
    int q = ((int)i & 31) * 4;
    int r = row[e], c = col[e];
    const float4 v = *(const float4*)(h + (size_t)r * HID + q);
    float* dst = agg + (size_t)c * HID + q;
    unsafeAtomicAdd(dst + 0, v.x);
    unsafeAtomicAdd(dst + 1, v.y);
    unsafeAtomicAdd(dst + 2, v.z);
    unsafeAtomicAdd(dst + 3, v.w);
}

// --- Node MLP + residual + LayerNorm. One wave per 16-node tile, 4 waves/block.
__global__ __launch_bounds__(128)
void node_mlp_kernel(float* __restrict__ h, half_t* __restrict__ h16,
                     const float* __restrict__ agg,
                     const half_t* __restrict__ w1, const float* __restrict__ b1,
                     const half_t* __restrict__ w2, const float* __restrict__ b2,
                     const float* __restrict__ lng, const float* __restrict__ lnb) {
    __shared__ __align__(16) half_t Az[4][16][2 * HID];
    __shared__ __align__(16) half_t Z1[4][16][HID];
    __shared__ __align__(16) float  HO[4][16][HID];
    __shared__ float MU[4][16][2];

    int wave = threadIdx.x >> 5, lane = threadIdx.x & 31;
    int m0 = (blockIdx.x * 4 + wave) * 16;

    for (int idx = lane; idx < 16 * 32; idx += 32) {
        int r = idx >> 5, ch = idx & 31;
        int node = m0 + r; if (node >= NN) node = NN - 1;
        const float* src = (ch < 16) ? (h   + (size_t)node * HID + ch * 8)
                                     : (agg + (size_t)node * HID + (ch - 16) * 8);
#pragma unroll
        for (int q = 0; q < 8; q++) Az[wave][r][ch * 8 + q] = (half_t)src[q];
    }
    __syncthreads();

    for (int nt = 0; nt < 8; nt++) {
        v8f acc = {};
#pragma unroll
        for (int kt = 0; kt < 8; kt++) {
            v16h a = a_frag_lds(&Az[wave][0][0], 2 * HID, kt, lane);
            v16h b = b_frag(w1, 8, nt, kt, lane);
            acc = wmma_f16(a, b, acc);
        }
        int n = nt * 16 + (lane & 15);
        float bias = b1[n];
#pragma unroll
        for (int r = 0; r < 8; r++) {
            int m = r + ((lane >> 4) << 3);
            Z1[wave][m][n] = (half_t)silu_f(acc[r] + bias);
        }
    }
    __syncthreads();

    for (int nt = 0; nt < 8; nt++) {
        v8f acc = {};
#pragma unroll
        for (int kt = 0; kt < 4; kt++) {
            v16h a = a_frag_lds(&Z1[wave][0][0], HID, kt, lane);
            v16h b = b_frag(w2, 4, nt, kt, lane);
            acc = wmma_f16(a, b, acc);
        }
        int n = nt * 16 + (lane & 15);
        float bias = b2[n];
#pragma unroll
        for (int r = 0; r < 8; r++) {
            int m = r + ((lane >> 4) << 3);
            int node = m0 + m; int nc = (node < NN) ? node : (NN - 1);
            HO[wave][m][n] = h[(size_t)nc * HID + n] + acc[r] + bias;
        }
    }
    __syncthreads();

    if (lane < 16) {
        float s = 0.0f, ss = 0.0f;
#pragma unroll 8
        for (int c = 0; c < HID; c++) { float v = HO[wave][lane][c]; s += v; ss += v * v; }
        float mu  = s * (1.0f / HID);
        float var = ss * (1.0f / HID) - mu * mu;
        MU[wave][lane][0] = mu;
        MU[wave][lane][1] = rsqrtf(var + 1e-5f);
    }
    __syncthreads();
    {
        int m = lane & 15, hs = lane >> 4;
        int node = m0 + m;
        float mu = MU[wave][m][0], rstd = MU[wave][m][1];
        for (int c = hs * 64; c < hs * 64 + 64; c++) {
            float v = (HO[wave][m][c] - mu) * rstd * lng[c] + lnb[c];
            if (node < NN) {
                h  [(size_t)node * HID + c] = v;
                h16[(size_t)node * HID + c] = (half_t)v;
            }
        }
    }
}

// --- Edge MLP (dominant kernel). 4 waves/block, TPW tiles/wave,
//     LDS-resident weights, async double-buffered gathers.
__global__ __launch_bounds__(128)
void edge_mlp_kernel(const half_t* __restrict__ h16,
                     const int* __restrict__ row, const int* __restrict__ col,
                     float* __restrict__ ea,
                     const half_t* __restrict__ w1g, const float* __restrict__ b1,
                     const half_t* __restrict__ w2g, const float* __restrict__ b2) {
    __shared__ __align__(32) half_t W1[KE * EP];          // 40 KB, fragment order
    __shared__ __align__(32) half_t W2[EP * EP];          //  8 KB
    __shared__ __align__(16) half_t A[4][2][16][KE];      // 80 KB, double buffer
    __shared__ __align__(16) half_t EZ[4][16][EP];        //  8 KB

    int tid = threadIdx.x, wave = tid >> 5, lane = tid & 31;

    // stage weights once per block (shared by all 4 waves)
    for (int i = tid; i < (KE * EP) / 8; i += 128)
        *(v8h*)&W1[i * 8] = *(const v8h*)(w1g + (size_t)i * 8);
    for (int i = tid; i < (EP * EP) / 8; i += 128)
        *(v8h*)&W2[i * 8] = *(const v8h*)(w2g + (size_t)i * 8);
    __syncthreads();

    long base = ((long)blockIdx.x * 4 + wave) * (16 * TPW);

    // issue per-lane async gathers of h16[row]/h16[col] for one tile
    auto issue_gather = [&](int buf, long e0t) {
        for (int idx = lane; idx < 512; idx += 32) {       // 16 async b128 / lane-loop
            int r = idx >> 5;                              // edge row 0..15
            int c = idx & 31;                              // 16 chunks x 2 sources
            long e = e0t + r;
            int nd = (c < 16) ? row[e] : col[e];
            int hc = (c & 15) * 8;                         // half offset in [0,128)
            int dst = (c < 16) ? hc : (128 + hc);
            async_b128((unsigned)(uintptr_t)&A[wave][buf][r][dst],
                       h16 + (size_t)nd * HID + hc);
        }
    };

    issue_gather(0, base);

    for (int t = 0; t < TPW; t++) {
        int cur = t & 1;
        long e0t = base + (long)t * 16;

        // stage ea columns (f32 -> f16) for tile t; prefetch next tile's ea
        for (int idx = lane; idx < 128; idx += 32) {
            int r = idx >> 3, ch = idx & 7;
            const float* src = ea + (size_t)(e0t + r) * EP + ch * 8;
#pragma unroll
            for (int q = 0; q < 8; q++) A[wave][cur][r][256 + ch * 8 + q] = (half_t)src[q];
        }
        if (t + 1 < TPW) {
            if (lane < 16)
                __builtin_prefetch(ea + (size_t)(e0t + 16 + lane) * EP, 0, 1);
            issue_gather(cur ^ 1, e0t + 16);               // overlap with compute
            wait_async_le16();                             // tile t's 16 ops done
        } else {
            wait_async_0();
        }
        __syncthreads();

        // GEMM1: [16,320]@[320,64] + silu   (weights from LDS)
        for (int nt = 0; nt < 4; nt++) {
            v8f acc = {};
#pragma unroll
            for (int kt = 0; kt < 10; kt++) {
                v16h a = a_frag_lds(&A[wave][cur][0][0], KE, kt, lane);
                v16h b = b_frag(W1, 10, nt, kt, lane);
                acc = wmma_f16(a, b, acc);
            }
            int n = nt * 16 + (lane & 15);
            float bias = (n < EDIM) ? b1[n] : 0.0f;
#pragma unroll
            for (int r = 0; r < 8; r++) {
                int m = r + ((lane >> 4) << 3);
                EZ[wave][m][n] = (half_t)silu_f(acc[r] + bias);
            }
        }
        __syncthreads();

        // GEMM2: [16,64]@[64,64] + residual into f32 ea
        for (int nt = 0; nt < 4; nt++) {
            v8f acc = {};
#pragma unroll
            for (int kt = 0; kt < 2; kt++) {
                v16h a = a_frag_lds(&EZ[wave][0][0], EP, kt, lane);
                v16h b = b_frag(W2, 2, nt, kt, lane);
                acc = wmma_f16(a, b, acc);
            }
            int n = nt * 16 + (lane & 15);
            float bias = (n < EDIM) ? b2[n] : 0.0f;
#pragma unroll
            for (int r = 0; r < 8; r++) {
                int m = r + ((lane >> 4) << 3);
                float* p = ea + (size_t)(e0t + m) * EP + n;
                *p = *p + acc[r] + bias;
            }
        }
        __syncthreads();
    }
}

// --- Force readout + scatter into forces[col]
__global__ __launch_bounds__(128)
void force_kernel(const float* __restrict__ ea,
                  const half_t* __restrict__ w1, const float* __restrict__ b1,
                  const float* __restrict__ w2, const float* __restrict__ b2,
                  const int* __restrict__ row, const int* __restrict__ col,
                  const float* __restrict__ pos, float* __restrict__ out) {
    __shared__ __align__(16) half_t A [4][16][EP];
    __shared__ __align__(16) half_t FZ[4][16][EP];
    __shared__ float FM[4][16];

    int wave = threadIdx.x >> 5, lane = threadIdx.x & 31;
    long e0 = ((long)blockIdx.x * 4 + wave) * 16;

    for (int idx = lane; idx < 16 * 8; idx += 32) {
        int r = idx >> 3, ch = idx & 7;
        const float* src = ea + (size_t)(e0 + r) * EP + ch * 8;
#pragma unroll
        for (int q = 0; q < 8; q++) A[wave][r][ch * 8 + q] = (half_t)src[q];
    }
    __syncthreads();

    for (int nt = 0; nt < 4; nt++) {
        v8f acc = {};
#pragma unroll
        for (int kt = 0; kt < 2; kt++) {
            v16h a = a_frag_lds(&A[wave][0][0], EP, kt, lane);
            v16h b = b_frag(w1, 2, nt, kt, lane);
            acc = wmma_f16(a, b, acc);
        }
        int n = nt * 16 + (lane & 15);
        float bias = b1[n];
#pragma unroll
        for (int r = 0; r < 8; r++) {
            int m = r + ((lane >> 4) << 3);
            FZ[wave][m][n] = (half_t)silu_f(acc[r] + bias);
        }
    }
    __syncthreads();

    if (lane < 16) {
        float fm = b2[0];
#pragma unroll 8
        for (int k = 0; k < EP; k++) fm += (float)FZ[wave][lane][k] * w2[k];
        FM[wave][lane] = fm;
    }
    __syncthreads();

    if (lane < 16) {
        long e = e0 + lane;
        int rr = row[e], cc = col[e];
        float dx = pos[(size_t)rr * 3 + 0] - pos[(size_t)cc * 3 + 0];
        float dy = pos[(size_t)rr * 3 + 1] - pos[(size_t)cc * 3 + 1];
        float dz = pos[(size_t)rr * 3 + 2] - pos[(size_t)cc * 3 + 2];
        float inv = 1.0f / (sqrtf(dx * dx + dy * dy + dz * dz) + 1e-8f);
        float s = FM[wave][lane] * inv;     // contrib = fm * diff / (|diff|+eps)
        unsafeAtomicAdd(&out[(size_t)cc * 3 + 0], s * dx);
        unsafeAtomicAdd(&out[(size_t)cc * 3 + 1], s * dy);
        unsafeAtomicAdd(&out[(size_t)cc * 3 + 2], s * dz);
    }
}

extern "C" void kernel_launch(void* const* d_in, const int* in_sizes, int n_in,
                              void* d_out, int out_size, void* d_ws, size_t ws_size,
                              hipStream_t stream) {
    (void)in_sizes; (void)n_in; (void)out_size; (void)ws_size;

    const float* x     = (const float*)d_in[0];
    const float* pos   = (const float*)d_in[1];
    const float* eattr = (const float*)d_in[2];
    const int*   eidx  = (const int*)  d_in[3];
    const float* emb_w = (const float*)d_in[4];
    const float* emb_b = (const float*)d_in[5];
    const float* nm_w1 = (const float*)d_in[6];
    const float* nm_b1 = (const float*)d_in[7];
    const float* nm_w2 = (const float*)d_in[8];
    const float* nm_b2 = (const float*)d_in[9];
    const float* ln_g  = (const float*)d_in[10];
    const float* ln_b  = (const float*)d_in[11];
    const float* em_w1 = (const float*)d_in[12];
    const float* em_b1 = (const float*)d_in[13];
    const float* em_w2 = (const float*)d_in[14];
    const float* em_b2 = (const float*)d_in[15];
    const float* fr_w1 = (const float*)d_in[16];
    const float* fr_b1 = (const float*)d_in[17];
    const float* fr_w2 = (const float*)d_in[18];
    const float* fr_b2 = (const float*)d_in[19];
    const int* row = eidx;
    const int* col = eidx + NE;

    char* ws = (char*)d_ws;
    size_t off = 0;
    auto take = [&](size_t bytes) -> char* {
        char* p = ws + off;
        off = (off + bytes + 255) & ~(size_t)255;
        return p;
    };
    float*  h   = (float*) take((size_t)NN * HID * 4);
    half_t* h16 = (half_t*)take((size_t)NN * HID * 2);
    float*  agg = (float*) take((size_t)NN * HID * 4);
    float*  ea  = (float*) take((size_t)NE * EP * 4);
    half_t* w1n = (half_t*)take((size_t)NL * 256 * 128 * 2);
    half_t* w2n = (half_t*)take((size_t)NL * 128 * 128 * 2);
    half_t* w1e = (half_t*)take((size_t)NL * KE * EP * 2);
    half_t* w2e = (half_t*)take((size_t)NL * EP * EP * 2);
    half_t* fw1 = (half_t*)take((size_t)EP * EP * 2);

    for (int l = 0; l < NL; l++) {
        pack_b_kernel<<<(256 * 128 + 255) / 256, 256, 0, stream>>>(
            nm_w1 + (size_t)l * 256 * 128, w1n + (size_t)l * 256 * 128, 256, 128, 256, 128);
        pack_b_kernel<<<(128 * 128 + 255) / 256, 256, 0, stream>>>(
            nm_w2 + (size_t)l * 128 * 128, w2n + (size_t)l * 128 * 128, 128, 128, 128, 128);
        pack_b_kernel<<<(KE * EP + 255) / 256, 256, 0, stream>>>(
            em_w1 + (size_t)l * 306 * 50, w1e + (size_t)l * KE * EP, 306, 50, KE, EP);
        pack_b_kernel<<<(EP * EP + 255) / 256, 256, 0, stream>>>(
            em_w2 + (size_t)l * 50 * 50, w2e + (size_t)l * EP * EP, 50, 50, EP, EP);
    }
    pack_b_kernel<<<(EP * EP + 255) / 256, 256, 0, stream>>>(fr_w1, fw1, 50, 64, EP, EP);

    embed_kernel<<<((long)NN * HID + 255) / 256, 256, 0, stream>>>(x, emb_w, emb_b, h, h16);
    smear_kernel<<<((long)NE * EP + 255) / 256, 256, 0, stream>>>(eattr, ea);

    for (int l = 0; l < NL; l++) {
        zero_kernel<<<((long)NN * HID + 255) / 256, 256, 0, stream>>>(agg, (long)NN * HID);
        scatter_kernel<<<((long)NE * 32 + 255) / 256, 256, 0, stream>>>(h, row, col, agg);
        node_mlp_kernel<<<(NN + 63) / 64, 128, 0, stream>>>(
            h, h16, agg,
            w1n + (size_t)l * 256 * 128, nm_b1 + (size_t)l * HID,
            w2n + (size_t)l * 128 * 128, nm_b2 + (size_t)l * HID,
            ln_g + (size_t)l * HID, ln_b + (size_t)l * HID);
        edge_mlp_kernel<<<NE / (64 * TPW), 128, 0, stream>>>(
            h16, row, col, ea,
            w1e + (size_t)l * KE * EP, em_b1 + (size_t)l * EDIM,
            w2e + (size_t)l * EP * EP, em_b2 + (size_t)l * EDIM);
    }

    zero_kernel<<<((long)NN * 3 + 255) / 256, 256, 0, stream>>>((float*)d_out, (long)NN * 3);
    force_kernel<<<NE / 64, 128, 0, stream>>>(ea, fw1, fr_b1, fr_w2, fr_b2,
                                              row, col, pos, (float*)d_out);
}
